// RelativeAttention_84877143704111
// MI455X (gfx1250) — matrix operand order repaired and checked by reading, not compiled, not run on hardware
//
#include <hip/hip_runtime.h>
#include <hip/hip_bf16.h>

typedef _Float16 v8h  __attribute__((ext_vector_type(8)));
typedef _Float16 v16h __attribute__((ext_vector_type(16)));
typedef float    v8f  __attribute__((ext_vector_type(8)));

#define WMMA_F16(A,B,C) __builtin_amdgcn_wmma_f32_16x16x32_f16(false,(A),false,(B),(short)0,(C),false,false)

__device__ __forceinline__ v16h join16(v8h lo, v8h hi) {
  v16h r;
#pragma unroll
  for (int i = 0; i < 8; ++i) { r[i] = lo[i]; r[i + 8] = hi[i]; }
  return r;
}

// A-matrix 16x32 f16 fragment from row-major [16 x stride] source.
// ISA layout: lane m=l%16 holds row m; g=l/16: V0-3 = K 8g..8g+7, V4-7 = K 16+8g..23+8g.
__device__ __forceinline__ v16h load_a_frag(const _Float16* __restrict__ base, int stride) {
  const int lane = threadIdx.x & 31;
  const int m = lane & 15, g = lane >> 4;
  const _Float16* p = base + (size_t)m * stride;
  v8h lo = *(const v8h*)(p + 8 * g);
  v8h hi = *(const v8h*)(p + 16 + 8 * g);
  return join16(lo, hi);
}

// B-matrix 32x16 f16 fragment from *transposed* (row-major [N x stride]) source:
// lane n=l%16 holds column n; g=l/16: elements 0-7 = K 8g..8g+7, 8-15 = K 16+8g..23+8g.
__device__ __forceinline__ v16h load_b_frag(const _Float16* __restrict__ baseT, int stride) {
  const int lane = threadIdx.x & 31;
  const int n = lane & 15, g = lane >> 4;
  const _Float16* p = baseT + (size_t)n * stride;
  v8h lo = *(const v8h*)(p + 8 * g);
  v8h hi = *(const v8h*)(p + 16 + 8 * g);
  return join16(lo, hi);
}

// ---------------------------------------------------------------- precompute
__global__ void cvt_f32_to_f16(const float* __restrict__ in, _Float16* __restrict__ out, int n) {
  int i = blockIdx.x * blockDim.x + threadIdx.x;
  if (i < n) out[i] = (_Float16)in[i];
}

// in [rows][cols] f32 -> out [cols][rows] f16
__global__ void transpose_to_f16(const float* __restrict__ in, _Float16* __restrict__ out,
                                 int rows, int cols) {
  int i = blockIdx.x * blockDim.x + threadIdx.x;
  if (i < rows * cols) {
    int r = i / cols, c = i - r * cols;
    out[(size_t)c * rows + r] = (_Float16)in[i];
  }
}

// rbh[h][p] = bias_table[relative_index[p]*8 + h], f16.  16 MB, lives in L2.
__global__ void build_rel_bias(const int* __restrict__ relidx, const float* __restrict__ table,
                               _Float16* __restrict__ rbh) {
  int i = blockIdx.x * blockDim.x + threadIdx.x;
  if (i < 8 * 1024 * 1024) {
    int h = i >> 20, p = i & ((1 << 20) - 1);
    rbh[i] = (_Float16)table[relidx[p] * 8 + h];
  }
}

// ---------------------------------------------------------------- QKV GEMM
// C[32768,1536] = xh[32768,512] @ w_qkv ; scattered as f16 into Q[bh][n][d],
// K[bh][n][d], Vt[bh][d][n].  Block 256 thr = 8 waves (4x2), tile 128x64,
// wave tile 32x32 (2x2 WMMA tiles), K loop 512/32.
__global__ void qkv_gemm(const _Float16* __restrict__ xh, const _Float16* __restrict__ wT,
                         _Float16* __restrict__ Qb, _Float16* __restrict__ Kb,
                         _Float16* __restrict__ Vt) {
  const int lane = threadIdx.x & 31;
  const int wave = threadIdx.x >> 5;
  const int wm = wave >> 1, wn = wave & 1;
  const int mb = blockIdx.y * 128 + wm * 32;
  const int nb = blockIdx.x * 64 + wn * 32;

  v8f c00 = {}, c01 = {}, c10 = {}, c11 = {};
  for (int kk = 0; kk < 512; kk += 32) {
    v16h a0 = load_a_frag(xh + (size_t)mb * 512 + kk, 512);
    v16h a1 = load_a_frag(xh + (size_t)(mb + 16) * 512 + kk, 512);
    v16h b0 = load_b_frag(wT + (size_t)nb * 512 + kk, 512);
    v16h b1 = load_b_frag(wT + (size_t)(nb + 16) * 512 + kk, 512);
    c00 = WMMA_F16(a0, b0, c00);
    c01 = WMMA_F16(a0, b1, c01);
    c10 = WMMA_F16(a1, b0, c10);
    c11 = WMMA_F16(a1, b1, c11);
  }
  const int g = lane >> 4, ln = lane & 15;
  v8f cs[2][2] = {{c00, c01}, {c10, c11}};
#pragma unroll
  for (int tm = 0; tm < 2; ++tm)
#pragma unroll
    for (int tn = 0; tn < 2; ++tn) {
      int n3 = nb + tn * 16 + ln;
      int which = n3 >> 9;           // 0=q 1=k 2=v
      int hh = (n3 & 511) >> 6;
      int d = n3 & 63;
#pragma unroll
      for (int r = 0; r < 8; ++r) {
        int m = mb + tm * 16 + r + 8 * g;
        int b = m >> 10, i = m & 1023;
        int bh = b * 8 + hh;
        _Float16 v = (_Float16)cs[tm][tn][r];
        if (which == 0)      Qb[((size_t)bh * 1024 + i) * 64 + d] = v;
        else if (which == 1) Kb[((size_t)bh * 1024 + i) * 64 + d] = v;
        else                 Vt[((size_t)bh * 64 + d) * 1024 + i] = v;
      }
    }
}

// ---------------------------------------------------------------- flash attention
// Grid (8 row-blocks, 256 bh). 8 waves/block; each wave owns 16 query rows,
// streams keys in tiles of 32 with online softmax. 8 WMMA per key tile.
__global__ void flash_attn(const _Float16* __restrict__ Qb, const _Float16* __restrict__ Kb,
                           const _Float16* __restrict__ Vt, const _Float16* __restrict__ rbh,
                           _Float16* __restrict__ attn) {
  __shared__ _Float16 ptile[8][16 * 32];
  const int lane = threadIdx.x & 31;
  const int wave = threadIdx.x >> 5;
  const int g = lane >> 4, ln = lane & 15;
  const int bh = blockIdx.y;
  const int b = bh >> 3, h = bh & 7;
  const int i0 = blockIdx.x * 128 + wave * 16;

  const _Float16* qbase = Qb + ((size_t)bh * 1024 + i0) * 64;
  const v16h aq0 = load_a_frag(qbase, 64);       // d 0..31
  const v16h aq1 = load_a_frag(qbase + 32, 64);  // d 32..63

  float mrow[8], lrow[8];
  v8f acc[4] = {{}, {}, {}, {}};
#pragma unroll
  for (int r = 0; r < 8; ++r) { mrow[r] = -1e30f; lrow[r] = 0.0f; }

  const float scale = 0.125f;  // 64^-0.5
  const _Float16* rb = rbh + ((size_t)h << 20);
  const _Float16* vbase = Vt + (size_t)bh * 64 * 1024;

  for (int j0 = 0; j0 < 1024; j0 += 32) {
    // ---- S = q @ k^T for 16 rows x 32 keys
    const _Float16* kb0 = Kb + ((size_t)bh * 1024 + j0) * 64;
    v16h bk00 = load_b_frag(kb0, 64);
    v16h bk01 = load_b_frag(kb0 + 32, 64);
    v16h bk10 = load_b_frag(kb0 + 16 * 64, 64);
    v16h bk11 = load_b_frag(kb0 + 16 * 64 + 32, 64);
    v8f s0 = {}, s1 = {};
    s0 = WMMA_F16(aq0, bk00, s0);
    s0 = WMMA_F16(aq1, bk01, s0);
    s1 = WMMA_F16(aq0, bk10, s1);
    s1 = WMMA_F16(aq1, bk11, s1);

    // ---- scale + relative bias + online softmax (per-row over 16 lanes)
#pragma unroll
    for (int r = 0; r < 8; ++r) {
      int i = i0 + r + 8 * g;
      float bb0 = (float)rb[(size_t)i * 1024 + j0 + ln];
      float bb1 = (float)rb[(size_t)i * 1024 + j0 + 16 + ln];
      float t0 = s0[r] * scale + bb0;
      float t1 = s1[r] * scale + bb1;
      float tmax = fmaxf(t0, t1);
      tmax = fmaxf(tmax, __shfl_xor(tmax, 1, 32));
      tmax = fmaxf(tmax, __shfl_xor(tmax, 2, 32));
      tmax = fmaxf(tmax, __shfl_xor(tmax, 4, 32));
      tmax = fmaxf(tmax, __shfl_xor(tmax, 8, 32));
      float mnew = fmaxf(mrow[r], tmax);
      float alpha = __expf(mrow[r] - mnew);
      mrow[r] = mnew;
      float p0 = __expf(t0 - mnew);
      float p1 = __expf(t1 - mnew);
      float rs = p0 + p1;
      rs += __shfl_xor(rs, 1, 32);
      rs += __shfl_xor(rs, 2, 32);
      rs += __shfl_xor(rs, 4, 32);
      rs += __shfl_xor(rs, 8, 32);
      lrow[r] = lrow[r] * alpha + rs;
#pragma unroll
      for (int t = 0; t < 4; ++t) acc[t][r] *= alpha;
      // C-layout -> row-major LDS tile (wave-private, 1 KB)
      ptile[wave][(r + 8 * g) * 32 + ln] = (_Float16)p0;
      ptile[wave][(r + 8 * g) * 32 + 16 + ln] = (_Float16)p1;
    }
    asm volatile("s_wait_dscnt 0x0" ::: "memory");

    // ---- A fragment of P from LDS (ds_load_b128 x2)
    v8h plo = *(const v8h*)&ptile[wave][ln * 32 + 8 * g];
    v8h phi = *(const v8h*)&ptile[wave][ln * 32 + 16 + 8 * g];
    v16h ap = join16(plo, phi);

    // ---- O += P @ V  (V stored transposed: lane=d, contiguous keys)
#pragma unroll
    for (int t = 0; t < 4; ++t) {
      const _Float16* p = vbase + (size_t)(t * 16 + ln) * 1024 + j0;
      v16h bv = join16(*(const v8h*)(p + 8 * g), *(const v8h*)(p + 16 + 8 * g));
      acc[t] = WMMA_F16(ap, bv, acc[t]);
    }
  }

  // ---- normalize, store as [b, n, h*64] f16 for the output projection
#pragma unroll
  for (int r = 0; r < 8; ++r) {
    float rl = 1.0f / lrow[r];
    int i = i0 + r + 8 * g;
    size_t orow = ((size_t)b * 1024 + i) * 512 + h * 64;
#pragma unroll
    for (int t = 0; t < 4; ++t) attn[orow + t * 16 + ln] = (_Float16)(acc[t][r] * rl);
  }
}

// ---------------------------------------------------------------- out projection
__global__ void out_gemm(const _Float16* __restrict__ ah, const _Float16* __restrict__ wT,
                         const float* __restrict__ bias, float* __restrict__ out) {
  const int lane = threadIdx.x & 31;
  const int wave = threadIdx.x >> 5;
  const int wm = wave >> 1, wn = wave & 1;
  const int mb = blockIdx.y * 128 + wm * 32;
  const int nb = blockIdx.x * 64 + wn * 32;

  v8f c00 = {}, c01 = {}, c10 = {}, c11 = {};
  for (int kk = 0; kk < 512; kk += 32) {
    v16h a0 = load_a_frag(ah + (size_t)mb * 512 + kk, 512);
    v16h a1 = load_a_frag(ah + (size_t)(mb + 16) * 512 + kk, 512);
    v16h b0 = load_b_frag(wT + (size_t)nb * 512 + kk, 512);
    v16h b1 = load_b_frag(wT + (size_t)(nb + 16) * 512 + kk, 512);
    c00 = WMMA_F16(a0, b0, c00);
    c01 = WMMA_F16(a0, b1, c01);
    c10 = WMMA_F16(a1, b0, c10);
    c11 = WMMA_F16(a1, b1, c11);
  }
  const int g = lane >> 4, ln = lane & 15;
  v8f cs[2][2] = {{c00, c01}, {c10, c11}};
#pragma unroll
  for (int tm = 0; tm < 2; ++tm)
#pragma unroll
    for (int tn = 0; tn < 2; ++tn) {
      int n = nb + tn * 16 + ln;
      float bv = bias[n];
#pragma unroll
      for (int r = 0; r < 8; ++r) {
        int m = mb + tm * 16 + r + 8 * g;
        out[(size_t)m * 512 + n] = cs[tm][tn][r] + bv;
      }
    }
}

// ---------------------------------------------------------------- launch
extern "C" void kernel_launch(void* const* d_in, const int* in_sizes, int n_in,
                              void* d_out, int out_size, void* d_ws, size_t ws_size,
                              hipStream_t stream) {
  const float* x          = (const float*)d_in[0];   // [32,1024,512]
  const float* w_qkv      = (const float*)d_in[1];   // [512,1536]
  const float* bias_table = (const float*)d_in[2];   // [45,8]
  const float* w_out      = (const float*)d_in[3];   // [512,512]
  const float* b_out      = (const float*)d_in[4];   // [512]
  const int*   relidx     = (const int*)d_in[5];     // [1024*1024]

  char* ws = (char*)d_ws;
  _Float16* xh    = (_Float16*)ws;               ws += (size_t)16777216 * 2;  // x f16
  _Float16* wqkvT = (_Float16*)ws;               ws += (size_t)786432 * 2;    // [1536][512]
  _Float16* woutT = (_Float16*)ws;               ws += (size_t)262144 * 2;    // [512][512]
  _Float16* rbh   = (_Float16*)ws;               ws += (size_t)8388608 * 2;   // [8][1024][1024]
  _Float16* Qb    = (_Float16*)ws;               ws += (size_t)16777216 * 2;  // [bh][n][64]
  _Float16* Kb    = (_Float16*)ws;               ws += (size_t)16777216 * 2;  // [bh][n][64]
  _Float16* Vt    = (_Float16*)ws;               ws += (size_t)16777216 * 2;  // [bh][64][n]
  _Float16* attn  = (_Float16*)ws;               ws += (size_t)16777216 * 2;  // [b*n][512]

  cvt_f32_to_f16<<<(16777216 + 255) / 256, 256, 0, stream>>>(x, xh, 16777216);
  transpose_to_f16<<<(786432 + 255) / 256, 256, 0, stream>>>(w_qkv, wqkvT, 512, 1536);
  transpose_to_f16<<<(262144 + 255) / 256, 256, 0, stream>>>(w_out, woutT, 512, 512);
  build_rel_bias<<<(8388608 + 255) / 256, 256, 0, stream>>>(relidx, bias_table, rbh);

  qkv_gemm<<<dim3(24, 256), 256, 0, stream>>>(xh, wqkvT, Qb, Kb, Vt);
  flash_attn<<<dim3(8, 256), 256, 0, stream>>>(Qb, Kb, Vt, rbh, attn);
  out_gemm<<<dim3(8, 256), 256, 0, stream>>>(attn, woutT, b_out, (float*)d_out);
}
